// BiMultiHeadAttention_38912403702272
// MI455X (gfx1250) — matrix operand order repaired and checked by reading, not compiled, not run on hardware
//
#include <hip/hip_runtime.h>
#include <hip/hip_fp16.h>

typedef __attribute__((ext_vector_type(16))) _Float16 v16h;
typedef __attribute__((ext_vector_type(8)))  float    v8f;
typedef unsigned int v4u __attribute__((ext_vector_type(4)));
typedef int          v8i __attribute__((ext_vector_type(8)));
typedef int          v4i __attribute__((ext_vector_type(4)));

#define HH 8
#define DD 32
#define EE 256
#define TT 16384
#define SS 256
#define BH 16
#define CLAMPV 50000.0f
#define NEGV   -9e15f

// att1 dynamic-LDS layout (kb stage MUST be at offset 0 for the TDM descriptor)
#define L_KBS   0                    // 256*32 f16            = 16384 B
#define L_SLOG  16384                // [16][257] f32         = 16448 B
#define L_SP    32832                // [16][264] f16         =  8448 B
#define L_AMF   41280                // [256] f32             =  1024 B
#define L_RMAX  42304                // [16][8] f32           =   512 B
#define L_RSUM  42816                // [16][8] f32           =   512 B
#define L_TOTAL 43328

// K index held by A-fragment element i (contiguous order k = 16*half + i, matching B)
__device__ __forceinline__ int a_k_of(int i, int half) {
  return half * 16 + i;
}

// monotone float<->uint encoding for atomic max on floats
__device__ __forceinline__ unsigned fenc(float f) {
  unsigned u = __float_as_uint(f);
  return (u & 0x80000000u) ? ~u : (u | 0x80000000u);
}
__device__ __forceinline__ float fdec(unsigned u) {
  u = (u & 0x80000000u) ? (u & 0x7FFFFFFFu) : ~u;
  return __uint_as_float(u);
}

// ---------------------------------------------------------------------------
// Projection GEMM: Y = (X @ W^T + bias) * scale, output in "shaped" f16 layout
// ---------------------------------------------------------------------------
__global__ void gemm_proj(const float* __restrict__ X, const float* __restrict__ W,
                          const float* __restrict__ bias, float scale,
                          _Float16* __restrict__ out, int nlenLog2)
{
  const int lane = threadIdx.x & 31;
  const int wv   = threadIdx.x >> 5;
  const int half = lane >> 4;
  const int l16  = lane & 15;
  const int mt   = blockIdx.x;
  const int nt   = blockIdx.y * 4 + wv;
  const long mrow = mt * 16 + l16;
  const int  ncol = nt * 16 + l16;
  const int  lmask = (1 << nlenLog2) - 1;
  v8f c = {};
  for (int kk = 0; kk < EE; kk += 32) {
    __builtin_prefetch(&X[mrow * EE + kk + 32], 0, 0);
    v16h a, b;
#pragma unroll
    for (int i = 0; i < 16; ++i)
      a[i] = (_Float16)X[mrow * EE + kk + a_k_of(i, half)];
#pragma unroll
    for (int i = 0; i < 16; ++i)
      b[i] = (_Float16)W[(long)ncol * EE + kk + half * 16 + i];
    c = __builtin_amdgcn_wmma_f32_16x16x32_f16(false, a, false, b, (short)0, c, false, false);
  }
  const int h = ncol >> 5, d = ncol & 31;
#pragma unroll
  for (int r = 0; r < 8; ++r) {
    int row = mt * 16 + r + 8 * half;
    int bb = row >> nlenLog2, pos = row & lmask;
    float v = (c[r] + bias[ncol]) * scale;
    out[(((long)(bb * HH + h) << nlenLog2) + pos) * DD + d] = (_Float16)v;
  }
}

// ---------------------------------------------------------------------------
// Pass 1: v->l attention. One block per (t-tile of 16, bh). 128 threads.
// K slab for this bh staged into LDS via TDM tensor_load_to_lds.
// ---------------------------------------------------------------------------
__global__ void att1(const _Float16* __restrict__ qb, const _Float16* __restrict__ kb,
                     const _Float16* __restrict__ vlb, const int* __restrict__ maskl,
                     float* __restrict__ outv, unsigned* __restrict__ cmax)
{
  extern __shared__ char smem[];
  _Float16* kbs            = (_Float16*)(smem + L_KBS);
  float (*slog)[SS + 1]    = (float (*)[SS + 1])(smem + L_SLOG);
  _Float16 (*sp)[SS + 8]   = (_Float16 (*)[SS + 8])(smem + L_SP);
  float* samf              = (float*)(smem + L_AMF);
  float (*rmax)[8]         = (float (*)[8])(smem + L_RMAX);
  float (*rsumA)[8]        = (float (*)[8])(smem + L_RSUM);

  const int lane = threadIdx.x & 31;
  const int wv   = threadIdx.x >> 5;
  const int half = lane >> 4;
  const int l16  = lane & 15;
  const int tt   = blockIdx.x;
  const int bh   = blockIdx.y;
  const int bidx = bh >> 3;

  // ---- stage K[bh] (256x32 f16 = 16KB) into LDS ----
  const _Float16* ksrc = kb + (long)bh * SS * DD;
#if __has_builtin(__builtin_amdgcn_tensor_load_to_lds) && __has_builtin(__builtin_amdgcn_s_wait_tensorcnt)
  if (wv == 0) {
    unsigned long long ga = (unsigned long long)(uintptr_t)ksrc;
    v4u g0;
    g0[0] = 1u;                                            // count=1, user mode
    g0[1] = 0u;                                            // lds_addr = 0
    g0[2] = (unsigned)(ga & 0xFFFFFFFFu);                  // global_addr lo
    g0[3] = (unsigned)((ga >> 32) & 0x01FFFFFFu) | 0x80000000u; // addr hi | type=2
    v8i g1;
    g1[0] = 0x00030000;    // wg_mask=0, data_size=3 (8B), no flags
    g1[1] = 0x08000000;    // tensor_dim0 = 2048 (bits 79:48, low16 here)
    g1[2] = 0x00010000;    // tensor_dim1 = 1
    g1[3] = 0x08000000;    // tile_dim0 = 2048 (bits 127:112)
    g1[4] = 0x00000001;    // tile_dim1 = 1
    g1[5] = 2048;          // tensor_dim0_stride = 2048
    g1[6] = 0; g1[7] = 0;
    v4i z4 = {0, 0, 0, 0};
#if defined(__clang_major__) && __clang_major__ >= 23
    v8i z8 = {0, 0, 0, 0, 0, 0, 0, 0};
    __builtin_amdgcn_tensor_load_to_lds(g0, g1, z4, z4, z8, 0);
#else
    __builtin_amdgcn_tensor_load_to_lds(g0, g1, z4, z4, 0);
#endif
    __builtin_amdgcn_s_wait_tensorcnt(0);
  }
#else
  for (int i = threadIdx.x; i < SS * DD / 4; i += 128)
    ((unsigned long long*)kbs)[i] = ((const unsigned long long*)ksrc)[i];
#endif

  // mask-add vector, computed once
  for (int s = threadIdx.x; s < SS; s += 128)
    samf[s] = (maskl[bidx * SS + s] == 0) ? NEGV : 1.0f;
  __syncthreads();

  // ---- logits: Q tile (A-frag) x 16 s-tiles of K (B-frags from LDS) ----
  v16h a;
#pragma unroll
  for (int i = 0; i < 16; ++i)
    a[i] = qb[((long)bh * TT + tt * 16 + l16) * DD + a_k_of(i, half)];

  for (int st = wv * 4; st < wv * 4 + 4; ++st) {
    v16h b;
#pragma unroll
    for (int i = 0; i < 16; ++i)
      b[i] = kbs[(st * 16 + l16) * DD + half * 16 + i];
    v8f c = {};
    c = __builtin_amdgcn_wmma_f32_16x16x32_f16(false, a, false, b, (short)0, c, false, false);
#pragma unroll
    for (int r = 0; r < 8; ++r) {
      float x = fminf(fmaxf(c[r], -CLAMPV), CLAMPV);
      slog[r + 8 * half][st * 16 + l16] = x;
    }
  }
  __syncthreads();

  // ---- per-column (s) max of clipped logits -> global atomic (for pass 2) ----
  {
    int s0 = threadIdx.x * 2;
#pragma unroll
    for (int s = s0; s < s0 + 2; ++s) {
      float m = -3.0e38f;
#pragma unroll
      for (int i = 0; i < 16; ++i) m = fmaxf(m, slog[i][s]);
      atomicMax(&cmax[bh * SS + s], fenc(m));
    }
  }
  __syncthreads();

  // ---- masked row softmax, parallel: row = tid&15, 32-col chunk = tid>>4 ----
  const int srow = threadIdx.x & 15;
  const int schunk = threadIdx.x >> 4;
  {
    float mx = -3.0e38f;
    for (int s = schunk * 32; s < schunk * 32 + 32; ++s) {
      float x = slog[srow][s] + samf[s];
      slog[srow][s] = x;
      mx = fmaxf(mx, x);
    }
    rmax[srow][schunk] = mx;
  }
  __syncthreads();
  if (schunk == 0) {
    float mx = rmax[srow][0];
#pragma unroll
    for (int j = 1; j < 8; ++j) mx = fmaxf(mx, rmax[srow][j]);
    rmax[srow][0] = mx;
  }
  __syncthreads();
  {
    float mx = rmax[srow][0];
    float sum = 0.f;
    for (int s = schunk * 32; s < schunk * 32 + 32; ++s) {
      float e = __expf(slog[srow][s] - mx);
      slog[srow][s] = e;
      sum += e;
    }
    rsumA[srow][schunk] = sum;
  }
  __syncthreads();
  if (schunk == 0) {
    float sm = 0.f;
#pragma unroll
    for (int j = 0; j < 8; ++j) sm += rsumA[srow][j];
    rsumA[srow][0] = 1.0f / sm;
  }
  __syncthreads();
  {
    float inv = rsumA[srow][0];
    for (int s = schunk * 32; s < schunk * 32 + 32; ++s)
      sp[srow][s] = (_Float16)(slog[srow][s] * inv);
  }
  __syncthreads();

  // ---- out_v tile [16 x 32] = P[16x256] @ VL[256x32] ----
  if (wv < 2) {
    v8f c = {};
    for (int kk = 0; kk < SS; kk += 32) {
      v16h pa, b;
#pragma unroll
      for (int i = 0; i < 16; ++i) pa[i] = sp[l16][kk + a_k_of(i, half)];
#pragma unroll
      for (int i = 0; i < 16; ++i)
        b[i] = vlb[((long)bh * SS + kk + half * 16 + i) * DD + wv * 16 + l16];
      c = __builtin_amdgcn_wmma_f32_16x16x32_f16(false, pa, false, b, (short)0, c, false, false);
    }
#pragma unroll
    for (int r = 0; r < 8; ++r)
      outv[((long)bh * TT + tt * 16 + r + 8 * half) * DD + wv * 16 + l16] = c[r];
  }
}

__global__ void cmax_init(unsigned* cm) { cm[blockIdx.x * 256 + threadIdx.x] = fenc(-3.0e38f); }
__global__ void cmax_dec(const unsigned* cm, float* cf) {
  int i = blockIdx.x * 256 + threadIdx.x;
  cf[i] = fdec(cm[i]);
}
__global__ void zero_f(float* p) { p[blockIdx.x * 256 + threadIdx.x] = 0.f; }

// ---------------------------------------------------------------------------
// Pass 2: l->v attention, T split into 8 chunks (grid.z) for occupancy.
// Partial numerators/denominators accumulated with global f32 atomics.
// ---------------------------------------------------------------------------
__global__ void att2(const _Float16* __restrict__ qb, const _Float16* __restrict__ kb,
                     const _Float16* __restrict__ vvb, const float* __restrict__ cmaxf,
                     float* __restrict__ outlAcc, float* __restrict__ rsumG)
{
  __shared__ _Float16 pb[16][40];
  __shared__ float srs[16];
  const int lane = threadIdx.x & 31;
  const int half = lane >> 4;
  const int l16  = lane & 15;
  const int st   = blockIdx.x;
  const int bh   = blockIdx.y;
  const int t0   = blockIdx.z * (TT / 16 / 8);   // 128 t-tiles per chunk
  if (threadIdx.x < 16) srs[threadIdx.x] = 0.f;

  float cm[8];
#pragma unroll
  for (int r = 0; r < 8; ++r) cm[r] = cmaxf[bh * SS + st * 16 + r + 8 * half];

  v16h ka;
#pragma unroll
  for (int i = 0; i < 16; ++i)
    ka[i] = kb[((long)bh * SS + st * 16 + l16) * DD + a_k_of(i, half)];

  float psum[8];
#pragma unroll
  for (int r = 0; r < 8; ++r) psum[r] = 0.f;
  v8f acc0 = {}, acc1 = {};
  __syncthreads();

  for (int tt = t0; tt < t0 + 128; ++tt) {
    v16h b;
#pragma unroll
    for (int i = 0; i < 16; ++i)
      b[i] = qb[((long)bh * TT + tt * 16 + l16) * DD + half * 16 + i];
    v8f c = {};
    c = __builtin_amdgcn_wmma_f32_16x16x32_f16(false, ka, false, b, (short)0, c, false, false);
    int ph = tt & 1;
#pragma unroll
    for (int r = 0; r < 8; ++r) {
      float x = fminf(fmaxf(c[r], -CLAMPV), CLAMPV);
      x = fminf(fmaxf(x - cm[r], -CLAMPV), CLAMPV);
      float p = __expf(x);
      psum[r] += p;
      pb[r + 8 * half][ph * 16 + l16] = (_Float16)p;
    }
    __syncthreads();
    if (ph == 1) {
      long tbase = (long)(tt - 1) * 16;
      v16h pa, b0, b1;
#pragma unroll
      for (int i = 0; i < 16; ++i) pa[i] = pb[l16][a_k_of(i, half)];
#pragma unroll
      for (int i = 0; i < 16; ++i) {
        long t = (long)bh * TT + tbase + half * 16 + i;
        b0[i] = vvb[t * DD + l16];
        b1[i] = vvb[t * DD + 16 + l16];
      }
      acc0 = __builtin_amdgcn_wmma_f32_16x16x32_f16(false, pa, false, b0, (short)0, acc0, false, false);
      acc1 = __builtin_amdgcn_wmma_f32_16x16x32_f16(false, pa, false, b1, (short)0, acc1, false, false);
      __syncthreads();
    }
  }

#pragma unroll
  for (int r = 0; r < 8; ++r) atomicAdd(&srs[r + 8 * half], psum[r]);
  __syncthreads();
  if (threadIdx.x < 16)
    atomicAdd(&rsumG[bh * SS + st * 16 + threadIdx.x], srs[threadIdx.x]);
#pragma unroll
  for (int r = 0; r < 8; ++r) {
    long row = (long)bh * SS + st * 16 + r + 8 * half;
    atomicAdd(&outlAcc[row * DD + l16],      acc0[r]);
    atomicAdd(&outlAcc[row * DD + 16 + l16], acc1[r]);
  }
}

__global__ void outl_fin(const float* __restrict__ acc, const float* __restrict__ rs,
                         float* __restrict__ outl)
{
  int i = blockIdx.x * 256 + threadIdx.x;       // 131072 elements
  outl[i] = acc[i] / rs[i >> 5];
}

// ---------------------------------------------------------------------------
// Column stats for the analytically collapsed FFT filter.
// ---------------------------------------------------------------------------
__global__ void colstats(const float* __restrict__ src, int L, float* __restrict__ stats)
{
  __shared__ float red[256];
  int idx = blockIdx.x;             // bh*32 + d
  int bh = idx >> 5, d = idx & 31;
  const float* base = src + (long)bh * L * DD + d;
  float s = 0.f;
  for (int p = threadIdx.x; p < L; p += 256) s += base[(long)p * DD];
  red[threadIdx.x] = s;
  __syncthreads();
  for (int o = 128; o > 0; o >>= 1) {
    if (threadIdx.x < o) red[threadIdx.x] += red[threadIdx.x + o];
    __syncthreads();
  }
  if (threadIdx.x == 0) {
    stats[idx]        = red[0];
    stats[512 + idx]  = base[0];
    stats[1024 + idx] = base[DD];
    stats[1536 + idx] = base[(long)(L - 2) * DD];
    stats[2048 + idx] = base[(long)(L - 1) * DD];
  }
}

// mean(fft_filter(x)) = S/L + (c2_w @ relu(c1_w @ yf0 + c1_b) + c2_b)/L
__global__ void filter_agg(const float* __restrict__ stats, int L,
                           const float* __restrict__ lap_w, const float* __restrict__ lap_b,
                           const float* __restrict__ fc1_w, const float* __restrict__ fc2_w,
                           const float* __restrict__ c1_w,  const float* __restrict__ c1_b,
                           const float* __restrict__ c2_w,  const float* __restrict__ c2_b,
                           float* __restrict__ agg)
{
  __shared__ float pooled[32], h1[32], yf0[64], z1[64], coefs;
  int b = blockIdx.x, tid = threadIdx.x;
  if (tid < 32) {
    float acc = 0.f;
    for (int c = 0; c < 32; ++c) {
      int i = b * 32 + c;
      float S  = stats[i];
      float e0 = stats[512 + i],  e1 = stats[1024 + i];
      float e2 = stats[1536 + i], e3 = stats[2048 + i];
      float t0 = S - e2 - e3, t1 = S - e0 - e3, t2 = S - e0 - e1;
      const float* w = lap_w + (tid * 32 + c) * 3;
      acc += w[0] * t0 + w[1] * t1 + w[2] * t2;
    }
    pooled[tid] = acc / (float)(L - 2) + lap_b[tid];
  }
  __syncthreads();
  if (tid < 32) {
    float acc = 0.f;
    for (int c = 0; c < 32; ++c) acc += fc1_w[tid * 32 + c] * pooled[c];
    h1[tid] = fmaxf(acc, 0.f);
  }
  __syncthreads();
  if (tid == 0) {
    float acc = 0.f;
    for (int c = 0; c < 32; ++c) acc += fc2_w[c] * h1[c];
    coefs = 1.0f / (1.0f + __expf(-acc));
  }
  __syncthreads();
  if (tid < 32) {
    float hw = (float)(L / 2);
    float g0 = __expf(-(hw * hw) / (2.0f * 7.0f * 7.0f));
    yf0[tid] = stats[b * 32 + tid] * (1.0f - coefs * g0);
    yf0[32 + tid] = 0.f;
  }
  __syncthreads();
  if (tid < 64) {
    float acc = c1_b[tid];
    for (int c = 0; c < 64; ++c) acc += c1_w[tid * 64 + c] * yf0[c];
    z1[tid] = fmaxf(acc, 0.f);
  }
  __syncthreads();
  if (tid < 32) {
    float acc = c2_b[tid];
    for (int c = 0; c < 64; ++c) acc += c2_w[tid * 64 + c] * z1[c];
    agg[b * 32 + tid] = (stats[b * 32 + tid] + acc) / (float)L;
  }
}

// ---------------------------------------------------------------------------
// Output GEMM: out = (shaped_src * agg, un-shaped) @ W^T + bias (f32)
// ---------------------------------------------------------------------------
__global__ void gemm_out(const float* __restrict__ srcShaped, const float* __restrict__ agg,
                         const float* __restrict__ W, const float* __restrict__ bias,
                         float* __restrict__ out, int nlenLog2)
{
  const int lane = threadIdx.x & 31;
  const int wv   = threadIdx.x >> 5;
  const int half = lane >> 4;
  const int l16  = lane & 15;
  const int mt   = blockIdx.x;
  const int nt   = blockIdx.y * 4 + wv;
  const int mrow = mt * 16 + l16;
  const int bb = mrow >> nlenLog2, pos = mrow & ((1 << nlenLog2) - 1);
  const int ncol = nt * 16 + l16;
  v8f c = {};
  for (int kk = 0; kk < EE; kk += 32) {
    int bhh = bb * HH + (kk >> 5);
    const float* srow = srcShaped + (((long)bhh << nlenLog2) + pos) * DD;
    const float* arow = agg + bhh * DD;
    __builtin_prefetch(&W[(long)ncol * EE + kk + 32], 0, 0);
    v16h a, b;
#pragma unroll
    for (int i = 0; i < 16; ++i) {
      int d = a_k_of(i, half);
      a[i] = (_Float16)(srow[d] * arow[d]);
    }
#pragma unroll
    for (int i = 0; i < 16; ++i)
      b[i] = (_Float16)W[(long)ncol * EE + kk + half * 16 + i];
    c = __builtin_amdgcn_wmma_f32_16x16x32_f16(false, a, false, b, (short)0, c, false, false);
  }
#pragma unroll
  for (int r = 0; r < 8; ++r) {
    int row = mt * 16 + r + 8 * half;
    out[(long)row * EE + ncol] = c[r] + bias[ncol];
  }
}

// ---------------------------------------------------------------------------
extern "C" void kernel_launch(void* const* d_in, const int* in_sizes, int n_in,
                              void* d_out, int out_size, void* d_ws, size_t ws_size,
                              hipStream_t stream) {
  (void)in_sizes; (void)n_in; (void)out_size; (void)ws_size;
  const float* v     = (const float*)d_in[0];
  const float* l     = (const float*)d_in[1];
  const int*   maskl = (const int*)d_in[2];
  const float* Wq  = (const float*)d_in[3];
  const float* Wk  = (const float*)d_in[4];
  const float* Wvv = (const float*)d_in[5];
  const float* Wvl = (const float*)d_in[6];
  const float* Wov = (const float*)d_in[7];
  const float* Wol = (const float*)d_in[8];
  const float* bq  = (const float*)d_in[9];
  const float* bk  = (const float*)d_in[10];
  const float* bvv = (const float*)d_in[11];
  const float* bvl = (const float*)d_in[12];
  const float* bov = (const float*)d_in[13];
  const float* bol = (const float*)d_in[14];

  char* ws = (char*)d_ws;
  _Float16* qb    = (_Float16*)(ws + 0);             // 16 MB
  _Float16* vvb   = (_Float16*)(ws + 16777216);      // 16 MB
  _Float16* kb    = (_Float16*)(ws + 33554432);      // 256 KB
  _Float16* vlb   = (_Float16*)(ws + 33816576);      // 256 KB
  float*    outv  = (float*)(ws + 34078720);         // 32 MB
  float*    outl  = (float*)(ws + 67633152);         // 512 KB
  unsigned* cmu   = (unsigned*)(ws + 68157440);      // 16 KB
  float*    cmf   = (float*)(ws + 68173824);         // 16 KB
  float*    stv   = (float*)(ws + 68190208);         // 10 KB
  float*    stl   = (float*)(ws + 68200448);         // 10 KB
  float*    fva   = (float*)(ws + 68210688);         // 2 KB
  float*    fra   = (float*)(ws + 68212736);         // 2 KB
  float*    olAcc = (float*)(ws + 68214784);         // 512 KB
  float*    rsumG = (float*)(ws + 68739072);         // 16 KB

  const float qscale = 0.17677669529663687f;         // D^-0.5

  cmax_init<<<16, 256, 0, stream>>>(cmu);
  zero_f<<<512, 256, 0, stream>>>(olAcc);            // 131072 floats
  zero_f<<<16, 256, 0, stream>>>(rsumG);             // 4096 floats

  gemm_proj<<<dim3(2048, 4), 128, 0, stream>>>(v, Wq,  bq,  qscale, qb,  14);
  gemm_proj<<<dim3(32,   4), 128, 0, stream>>>(l, Wk,  bk,  1.0f,   kb,  8);
  gemm_proj<<<dim3(2048, 4), 128, 0, stream>>>(v, Wvv, bvv, 1.0f,   vvb, 14);
  gemm_proj<<<dim3(32,   4), 128, 0, stream>>>(l, Wvl, bvl, 1.0f,   vlb, 8);

  att1<<<dim3(1024, 16), 128, L_TOTAL, stream>>>(qb, kb, vlb, maskl, outv, cmu);
  cmax_dec<<<16, 256, 0, stream>>>(cmu, cmf);
  att2<<<dim3(16, 16, 8), 32, 0, stream>>>(qb, kb, vvb, cmf, olAcc, rsumG);
  outl_fin<<<512, 256, 0, stream>>>(olAcc, rsumG, outl);

  colstats<<<512, 256, 0, stream>>>(outv, TT, stv);
  colstats<<<512, 256, 0, stream>>>(outl, SS, stl);

  filter_agg<<<16, 64, 0, stream>>>(stv, TT,
      (const float*)d_in[15], (const float*)d_in[16], (const float*)d_in[17],
      (const float*)d_in[18], (const float*)d_in[19], (const float*)d_in[20],
      (const float*)d_in[21], (const float*)d_in[22], fva);
  filter_agg<<<16, 64, 0, stream>>>(stl, SS,
      (const float*)d_in[23], (const float*)d_in[24], (const float*)d_in[25],
      (const float*)d_in[26], (const float*)d_in[27], (const float*)d_in[28],
      (const float*)d_in[29], (const float*)d_in[30], fra);

  float* out_v_final = (float*)d_out;                // [B,T,E]
  float* out_l_final = (float*)d_out + 8388608;      // [B,S,E]
  gemm_out<<<dim3(2048, 4), 128, 0, stream>>>(outv, fra, Wov, bov, out_v_final, 14);
  gemm_out<<<dim3(32,   4), 128, 0, stream>>>(outl, fva, Wol, bol, out_l_final, 8);
}
